// Sim_Block_88098369176506
// MI455X (gfx1250) — matrix operand order repaired and checked by reading, not compiled, not run on hardware
//
#include <hip/hip_runtime.h>
#include <hip/hip_bf16.h>

typedef __attribute__((ext_vector_type(16))) _Float16 v16h;
typedef __attribute__((ext_vector_type(8)))  _Float16 v8h;
typedef __attribute__((ext_vector_type(8)))  float    v8f;

#define HW           9216
#define CH           64
#define TOPK         7
#define NSEG         4
#define COLS_PER_SEG 2304          // 9216 / 4
#define CHUNK        32            // columns per LDS tile
#define NCHUNK       72            // 2304 / 32
#define ROWS_PER_BLK 256
#define LDS_STRIDE   35            // odd stride -> conflict-free per-lane row scan

// ---- workspace layout (bytes) ----
#define WS_XH_OFF    0                                   // 2*9216*64*2  = 2,359,296
#define WS_INORM_OFF 2359296                             // 2*9216*4    =    73,728
#define WS_PVAL_OFF  (2359296 + 73728)                   // 2*9216*28*4 = 2,064,384
#define WS_PIDX_OFF  (2359296 + 73728 + 2064384)         // 2*9216*28*4 = 2,064,384

// Stable descending top-k insertion: strict '>' keeps earlier-inserted (lower
// column index) first on ties, matching jax.lax.top_k tie-breaking.
__device__ __forceinline__ void topk_insert(float v, int col,
                                            float tv[TOPK], int ti[TOPK]) {
  if (v > tv[TOPK - 1]) {
    float cv = v; int ci = col;
#pragma unroll
    for (int s = 0; s < TOPK; ++s) {
      if (cv > tv[s]) {
        float t1 = tv[s]; int t2 = ti[s];
        tv[s] = cv; ti[s] = ci;
        cv = t1;  ci = t2;
      }
    }
  }
}

// -------- kernel 1: inverse norms + normalized f16 copy (channel-contiguous) --------
__global__ void __launch_bounds__(256)
prep_kernel(const float* __restrict__ x, _Float16* __restrict__ Xh,
            float* __restrict__ inorm) {
  const int gid = blockIdx.x * 256 + threadIdx.x;    // 0 .. 2*HW-1
  const int bb = gid / HW;
  const int p  = gid - bb * HW;
  const float* xb = x + (size_t)bb * CH * HW + p;
  float ss = 0.f;
#pragma unroll 8
  for (int c = 0; c < CH; ++c) { float v = xb[(size_t)c * HW]; ss += v * v; }
  const float rn = 1.0f / sqrtf(ss);
  inorm[gid] = rn;
  _Float16* xo = Xh + (size_t)gid * CH;
#pragma unroll 8
  for (int c = 0; c < CH; ++c) xo[c] = (_Float16)(xb[(size_t)c * HW] * rn);
}

// -------- kernel 2: WMMA similarity tiles + per-row partial top-7 over one column segment --------
__global__ void __launch_bounds__(256)
score_topk_kernel(const _Float16* __restrict__ Xh,
                  float* __restrict__ pvals, int* __restrict__ pidx) {
  __shared__ float sS[ROWS_PER_BLK * LDS_STRIDE];      // 35,840 B

  const int tid  = threadIdx.x;
  const int lane = tid & 31;
  const int wv   = tid >> 5;                           // 8 waves
  const int rb   = blockIdx.x;                         // 0..35
  const int seg  = blockIdx.y;                         // 0..3
  const int bb   = blockIdx.z;                         // 0..1
  const int rowBase = rb * ROWS_PER_BLK;
  const int segBase = seg * COLS_PER_SEG;
  const _Float16* Xb = Xh + (size_t)bb * HW * CH;

  float tv[TOPK]; int ti[TOPK];
#pragma unroll
  for (int s = 0; s < TOPK; ++s) { tv[s] = -3.4e38f; ti[s] = 0; }

  const bool lolane = (lane < 16);
  const int  l15 = lane & 15;
  const int  akb = lolane ? 0 : 16;                    // A: K base (contiguous, matches B)
  const int  bkb = lolane ? 0 : 16;                    // B: K base (contiguous, matches A)

  for (int chunk = 0; chunk < NCHUNK; ++chunk) {
    const int colBase = segBase + chunk * CHUNK;

    // prefetch next chunk's B rows (one row per lane, rows are 128B)
    if (chunk + 1 < NCHUNK) {
      const _Float16* pf = Xb + (size_t)(colBase + CHUNK + lane) * CH;
      __builtin_prefetch(pf, 0, 1);
    }

    // ---- WMMA: 256x32 score tile -> LDS (16 M-tiles x 2 N-tiles; 4 tiles/wave) ----
#pragma unroll
    for (int tmi = 0; tmi < 2; ++tmi) {
      const int tm = wv * 2 + tmi;
      const _Float16* pa = Xb + (size_t)(rowBase + tm * 16 + l15) * CH;
      union { v16h v; v8h h[2]; } A0, A1;
      A0.h[0] = *(const v8h*)(pa + akb);               // K =  akb..akb+7
      A0.h[1] = *(const v8h*)(pa + akb + 8);           // K =  akb+8..akb+15
      A1.h[0] = *(const v8h*)(pa + akb + 32);          // second K-half (+32)
      A1.h[1] = *(const v8h*)(pa + akb + 40);
#pragma unroll
      for (int tn = 0; tn < 2; ++tn) {
        const _Float16* pb = Xb + (size_t)(colBase + tn * 16 + l15) * CH;
        v16h B0 = *(const v16h*)(pb + bkb);            // K = bkb..bkb+15
        v16h B1 = *(const v16h*)(pb + bkb + 32);       // K = bkb+32..bkb+47
        v8f acc = {};
        acc = __builtin_amdgcn_wmma_f32_16x16x32_f16(false, A0.v, false, B0,
                                                     (short)0, acc, false, false);
        acc = __builtin_amdgcn_wmma_f32_16x16x32_f16(false, A1.v, false, B1,
                                                     (short)0, acc, false, false);
        // C/D layout: VGPR v -> (M = v + 8*hi_half, N = lane%16)
        const int r0 = tm * 16 + (lolane ? 0 : 8);
        const int c0 = tn * 16 + l15;
#pragma unroll
        for (int v = 0; v < 8; ++v)
          sS[(r0 + v) * LDS_STRIDE + c0] = acc[v];
      }
    }
    __syncthreads();

    // ---- per-thread scan of its own row (odd stride -> no bank conflicts) ----
    const float* row = &sS[tid * LDS_STRIDE];
#pragma unroll 8
    for (int j = 0; j < CHUNK; ++j)
      topk_insert(row[j], colBase + j, tv, ti);
    __syncthreads();
  }

  const size_t base = (size_t)(bb * HW + rowBase + tid) * (NSEG * TOPK) + seg * TOPK;
#pragma unroll
  for (int s = 0; s < TOPK; ++s) { pvals[base + s] = tv[s]; pidx[base + s] = ti[s]; }
}

// -------- kernel 3: merge partials, exact f32 values, conv(i,k) + bias + ReLU --------
__global__ void __launch_bounds__(256)
merge_conv_kernel(const float* __restrict__ x, const float* __restrict__ W,
                  const float* __restrict__ bias, const float* __restrict__ inorm,
                  const float* __restrict__ pvals, const int* __restrict__ pidx,
                  float* __restrict__ out) {
  const int tid = threadIdx.x;
  const int rb  = blockIdx.x;                          // 36
  const int og  = blockIdx.y;                          // 4 (16 output channels each)
  const int bb  = blockIdx.z;                          // 2
  const int m   = rb * 256 + tid;
  const int obase = og * 16;

  // merge 4 segment-local sorted top-7 lists (ascending column order preserved)
  float tv[TOPK]; int ti[TOPK];
#pragma unroll
  for (int s = 0; s < TOPK; ++s) { tv[s] = -3.4e38f; ti[s] = 0; }
  const size_t pbase = (size_t)(bb * HW + m) * (NSEG * TOPK);
#pragma unroll
  for (int c = 0; c < NSEG * TOPK; ++c)
    topk_insert(pvals[pbase + c], pidx[pbase + c], tv, ti);

  const float* xb = x + (size_t)bb * CH * HW;

  // exact f32 similarity values for the selected neighbors
  float sdot[TOPK];
#pragma unroll
  for (int s = 0; s < TOPK; ++s) sdot[s] = 0.f;
#pragma unroll 4
  for (int i = 0; i < CH; ++i) {
    const float* xr = xb + (size_t)i * HW;
    const float xm = xr[m];
#pragma unroll
    for (int s = 0; s < TOPK; ++s) sdot[s] += xm * xr[ti[s]];
  }
  const float inm = inorm[bb * HW + m];
  float val[TOPK];
#pragma unroll
  for (int s = 0; s < TOPK; ++s) val[s] = sdot[s] * inm * inorm[bb * HW + ti[s]];

  // out[o,m] = sum_{i,k} x[i, idx_k] * val_k * W[o,i,k]
  float acc[16];
#pragma unroll
  for (int oo = 0; oo < 16; ++oo) acc[oo] = 0.f;
  for (int i = 0; i < CH; ++i) {
    const float* xr = xb + (size_t)i * HW;
    float g[TOPK];
#pragma unroll
    for (int s = 0; s < TOPK; ++s) g[s] = xr[ti[s]] * val[s];
#pragma unroll
    for (int oo = 0; oo < 16; ++oo) {
      const float* Wo = W + ((size_t)(obase + oo) * CH + i) * TOPK;
      float t = g[0] * Wo[0] + g[1] * Wo[1] + g[2] * Wo[2] + g[3] * Wo[3]
              + g[4] * Wo[4] + g[5] * Wo[5] + g[6] * Wo[6];
      acc[oo] += t;
    }
  }
#pragma unroll
  for (int oo = 0; oo < 16; ++oo) {
    const float r = acc[oo] + bias[obase + oo];
    out[(size_t)(bb * CH + obase + oo) * HW + m] = fmaxf(r, 0.f);
  }
}

extern "C" void kernel_launch(void* const* d_in, const int* in_sizes, int n_in,
                              void* d_out, int out_size, void* d_ws, size_t ws_size,
                              hipStream_t stream) {
  (void)in_sizes; (void)n_in; (void)out_size; (void)ws_size;
  const float* x    = (const float*)d_in[0];   // [2,64,96,96] f32
  const float* W    = (const float*)d_in[1];   // [64,64,7]    f32
  const float* bias = (const float*)d_in[2];   // [64]         f32
  float* out = (float*)d_out;                  // [2,64,96,96] f32

  char* ws = (char*)d_ws;                      // needs ~6.6 MB
  _Float16* Xh   = (_Float16*)(ws + WS_XH_OFF);
  float*    inrm = (float*)   (ws + WS_INORM_OFF);
  float*    pval = (float*)   (ws + WS_PVAL_OFF);
  int*      pidx = (int*)     (ws + WS_PIDX_OFF);

  prep_kernel<<<dim3((2 * HW) / 256), dim3(256), 0, stream>>>(x, Xh, inrm);
  score_topk_kernel<<<dim3(36, 4, 2), dim3(256), 0, stream>>>(Xh, pval, pidx);
  merge_conv_kernel<<<dim3(36, 4, 2), dim3(256), 0, stream>>>(x, W, bias, inrm,
                                                              pval, pidx, out);
}
